// CudaFastWeightSumTwoPerformerLayer_15891378995465
// MI455X (gfx1250) — matrix operand order repaired and checked by reading, not compiled, not run on hardware
//
#include <hip/hip_runtime.h>
#include <hip/hip_bf16.h>
#include <stdint.h>

typedef __attribute__((ext_vector_type(16))) _Float16 v16h;
typedef __attribute__((ext_vector_type(8)))  float    v8f;
typedef unsigned int v4u __attribute__((ext_vector_type(4)));
typedef unsigned int v8u __attribute__((ext_vector_type(8)));
typedef _Float16 half_t;

#define LSEQ   256
#define BB     32
#define NH     8
#define DH     64
#define DMODEL 512
#define PM     256      // proj dim m
#define M2     512      // 2m feature dim
#define ROWS   8192     // LSEQ*BB
#define CHK    64

// ---------------- WMMA fragment helpers (per CDNA5 ISA §7.12.2) ----------------

__device__ __forceinline__ v8f zero8() {
  v8f z;
#pragma unroll
  for (int i = 0; i < 8; ++i) z[i] = 0.f;
  return z;
}

// A fragment, 16x32 f16, element (row, k) = src[row*ld + k]  (two 8-k runs per lane)
__device__ __forceinline__ v16h load_a_frag(const half_t* src, int ld) {
  int lane = threadIdx.x & 31;
  int row  = lane & 15;
  int hi   = lane >> 4;
  v16h a;
#pragma unroll
  for (int v = 0; v < 8; ++v) {
    int k = ((v < 4) ? (2 * v) : (2 * v + 8)) + hi * 8;
    a[2 * v]     = src[row * ld + k];
    a[2 * v + 1] = src[row * ld + k + 1];
  }
  return a;
}

// A fragment from transposed source: element (row, k) = src[k*ld + row]
__device__ __forceinline__ v16h load_a_frag_T(const half_t* src, int ld) {
  int lane = threadIdx.x & 31;
  int row  = lane & 15;
  int hi   = lane >> 4;
  v16h a;
#pragma unroll
  for (int v = 0; v < 8; ++v) {
    int k = ((v < 4) ? (2 * v) : (2 * v + 8)) + hi * 8;
    a[2 * v]     = src[k * ld + row];
    a[2 * v + 1] = src[(k + 1) * ld + row];
  }
  return a;
}

// B fragment from transposed (column-major) source: element (k, col) = src[col*ld + k]
// Per-lane footprint: two contiguous 16B runs -> vectorizes to b128 loads.
__device__ __forceinline__ v16h load_b_frag_T(const half_t* src, int ld) {
  int lane = threadIdx.x & 31;
  int col  = lane & 15;
  int hi   = lane >> 4;
  v16h b;
#pragma unroll
  for (int v = 0; v < 8; ++v) {
    int k = ((v < 4) ? (2 * v) : (2 * v + 8)) + hi * 8;
    b[2 * v]     = src[col * ld + k];
    b[2 * v + 1] = src[col * ld + k + 1];
  }
  return b;
}

// C/D 16x16 f32: element (row = v + hi*8, col = lane&15)
__device__ __forceinline__ void store_c(float* dst, int ld, const v8f& c) {
  int lane = threadIdx.x & 31;
  int col  = lane & 15;
  int hi   = lane >> 4;
#pragma unroll
  for (int v = 0; v < 8; ++v) dst[(v + hi * 8) * ld + col] = c[v];
}

__device__ __forceinline__ void store_c_h(half_t* dst, int ld, const v8f& c) {
  int lane = threadIdx.x & 31;
  int col  = lane & 15;
  int hi   = lane >> 4;
#pragma unroll
  for (int v = 0; v < 8; ++v) dst[(v + hi * 8) * ld + col] = (half_t)c[v];
}

// store C tile into a TRANSPOSED f16 array: element (row,col) -> dst[col*ld + row]
__device__ __forceinline__ void store_c_h_T(half_t* dst, int ld, const v8f& c) {
  int lane = threadIdx.x & 31;
  int col  = lane & 15;
  int hi   = lane >> 4;
#pragma unroll
  for (int v = 0; v < 8; ++v) dst[col * ld + v + hi * 8] = (half_t)c[v];
}

__device__ __forceinline__ v8f load_c(const float* src, int ld) {
  int lane = threadIdx.x & 31;
  int col  = lane & 15;
  int hi   = lane >> 4;
  v8f c;
#pragma unroll
  for (int v = 0; v < 8; ++v) c[v] = src[(v + hi * 8) * ld + col];
  return c;
}

__device__ __forceinline__ void atomic_add_c(float* dst, int ld, const v8f& c) {
  int lane = threadIdx.x & 31;
  int col  = lane & 15;
  int hi   = lane >> 4;
#pragma unroll
  for (int v = 0; v < 8; ++v) atomicAdd(&dst[(v + hi * 8) * ld + col], c[v]);
}

__device__ __forceinline__ v8f wmma32(v16h a, v16h b, v8f c) {
  return __builtin_amdgcn_wmma_f32_16x16x32_f16(false, a, false, b, (short)0, c,
                                                false, false);
}

// ---------------- Kernel 0a: f32 -> f16 cast ----------------
__global__ void k_cast_f16(const float* __restrict__ src, half_t* __restrict__ dst,
                           int n, float scale) {
  for (int i = blockIdx.x * blockDim.x + threadIdx.x; i < n;
       i += gridDim.x * blockDim.x)
    dst[i] = (half_t)(src[i] * scale);
}

// ---------------- Kernel 0b: f32 [R][C] -> f16 transposed [C][R] ----------------
__global__ void k_cast_t16(const float* __restrict__ src, half_t* __restrict__ dst,
                           int R, int C, float scale) {
  for (int i = blockIdx.x * blockDim.x + threadIdx.x; i < R * C;
       i += gridDim.x * blockDim.x) {
    int r = i / C, c = i - r * C;
    dst[(size_t)c * R + r] = (half_t)(src[i] * scale);
  }
}

// ---------------- Kernel 1: qkv = h @ w_qkv  (8192x512 @ 512x2048) ----------------
// B is pre-transposed: wqkvT[2048][512]. A tile staged via async global->LDS copy.
__global__ void k_gemm_qkv(const half_t* __restrict__ A, const half_t* __restrict__ BT,
                           half_t* __restrict__ C) {
  __shared__ half_t As[64 * 32];
  int rb   = blockIdx.x * 64;
  int cb   = blockIdx.y * 128;
  int wave = threadIdx.x >> 5;
  v8f acc[4];
#pragma unroll
  for (int i = 0; i < 4; ++i) acc[i] = zero8();

  // each thread owns one 16B chunk of the 64x32 f16 A tile
  int sr  = threadIdx.x >> 2;        // row 0..63
  int sc8 = (threadIdx.x & 3) * 8;   // half-offset 0/8/16/24
  unsigned ldsa = (unsigned)(uintptr_t)(As + sr * 32 + sc8);

  for (int kk = 0; kk < DMODEL; kk += 32) {
    {  // async copy 4KB A tile into LDS (one b128 per lane)
      unsigned long long ga =
          (unsigned long long)(uintptr_t)(A + (size_t)(rb + sr) * DMODEL + kk + sc8);
      asm volatile("global_load_async_to_lds_b128 %0, %1, off"
                   :: "v"(ldsa), "v"(ga) : "memory");
      asm volatile("s_wait_asynccnt 0" ::: "memory");
    }
    // speculative prefetch of next k-step's B fragment region
    __builtin_prefetch(BT + (size_t)(cb + wave * 16) * DMODEL + kk + 32, 0, 1);
    __syncthreads();
    v16h bf = load_b_frag_T(BT + (size_t)(cb + wave * 16) * DMODEL + kk, DMODEL);
#pragma unroll
    for (int rt = 0; rt < 4; ++rt) {
      v16h af = load_a_frag(As + rt * 16 * 32, 32);
      acc[rt] = wmma32(af, bf, acc[rt]);
    }
    __syncthreads();
  }
#pragma unroll
  for (int rt = 0; rt < 4; ++rt)
    store_c_h(C + (size_t)(rb + rt * 16) * 2048 + cb + wave * 16, 2048, acc[rt]);
}

// ---------------- Kernel 2: prime(q) -> qf16 [(b*8+h)*256+l][512] ----------------
// projT: [256][64] (transposed). Dynamic LDS: xp f32 [64][256].
__global__ void k_prime_q(const half_t* __restrict__ qkv, const half_t* __restrict__ projT,
                          half_t* __restrict__ qf) {
  extern __shared__ float s_xp[];
  int rb   = blockIdx.x * 64;
  int h    = blockIdx.y;
  int wave = threadIdx.x >> 5;
  const half_t* Ab = qkv + h * 256;  // q slice cols [h*256, h*256+64)

  v8f acc[2][4];
#pragma unroll
  for (int c = 0; c < 2; ++c)
#pragma unroll
    for (int r = 0; r < 4; ++r) acc[c][r] = zero8();

  for (int kk = 0; kk < DH; kk += 32) {
#pragma unroll
    for (int ct = 0; ct < 2; ++ct) {
      v16h bf = load_b_frag_T(projT + (size_t)((wave + ct * 8) * 16) * DH + kk, DH);
#pragma unroll
      for (int rt = 0; rt < 4; ++rt) {
        v16h af = load_a_frag(Ab + (size_t)(rb + rt * 16) * 2048 + kk, 2048);
        acc[ct][rt] = wmma32(af, bf, acc[ct][rt]);
      }
    }
  }
#pragma unroll
  for (int ct = 0; ct < 2; ++ct)
#pragma unroll
    for (int rt = 0; rt < 4; ++rt)
      store_c(s_xp + (rt * 16) * PM + (wave + ct * 8) * 16, PM, acc[ct][rt]);
  __syncthreads();

  // softmax over [xp, -xp] per row (-||x||^2/2 and rowmax cancel in normalization)
  int row = threadIdx.x >> 2;
  int j   = threadIdx.x & 3;
  float* xr = s_xp + row * PM;
  float mx = 0.f;
  for (int c = 0; c < 64; ++c) mx = fmaxf(mx, fabsf(xr[j * 64 + c]));
  mx = fmaxf(mx, __shfl_xor(mx, 1));
  mx = fmaxf(mx, __shfl_xor(mx, 2));
  float s = 0.f;
  for (int c = 0; c < 64; ++c) {
    float v = xr[j * 64 + c];
    s += __expf(v - mx) + __expf(-v - mx);
  }
  s += __shfl_xor(s, 1);
  s += __shfl_xor(s, 2);
  float inv = 1.f / s;
  int r = rb + row, l = r >> 5, b = r & 31;
  half_t* dst = qf + (size_t)((b * NH + h) * LSEQ + l) * M2;
  for (int c = 0; c < 64; c += 2) {  // packed b32 stores
    float v0 = xr[j * 64 + c], v1 = xr[j * 64 + c + 1];
    half_t p0 = (half_t)(__expf(v0 - mx) * inv), p1 = (half_t)(__expf(v1 - mx) * inv);
    half_t n0 = (half_t)(__expf(-v0 - mx) * inv), n1 = (half_t)(__expf(-v1 - mx) * inv);
    unsigned up = (unsigned)__builtin_bit_cast(unsigned short, p0) |
                  ((unsigned)__builtin_bit_cast(unsigned short, p1) << 16);
    unsigned un = (unsigned)__builtin_bit_cast(unsigned short, n0) |
                  ((unsigned)__builtin_bit_cast(unsigned short, n1) << 16);
    *(unsigned*)(dst + j * 64 + c)      = up;
    *(unsigned*)(dst + PM + j * 64 + c) = un;
  }
}

// ---------------- Kernel 3: kf = pi0*prime(k1) + pi1*prime(k2,0.7proj) ----------------
__global__ void k_prime_k(const half_t* __restrict__ qkv, const half_t* __restrict__ projT,
                          const half_t* __restrict__ proj2T, const float* __restrict__ pi0,
                          const float* __restrict__ pi1, half_t* __restrict__ kf) {
  extern __shared__ float s_xpk[];
  int rb   = blockIdx.x * 64;
  int h    = blockIdx.y;
  int wave = threadIdx.x >> 5;

  for (int ph = 0; ph < 2; ++ph) {
    const half_t* Ab = qkv + h * 256 + (ph == 0 ? 64 : 128);
    const half_t* Bp = (ph == 0) ? projT : proj2T;

    v8f acc[2][4];
#pragma unroll
    for (int c = 0; c < 2; ++c)
#pragma unroll
      for (int r = 0; r < 4; ++r) acc[c][r] = zero8();

    for (int kk = 0; kk < DH; kk += 32) {
#pragma unroll
      for (int ct = 0; ct < 2; ++ct) {
        v16h bf = load_b_frag_T(Bp + (size_t)((wave + ct * 8) * 16) * DH + kk, DH);
#pragma unroll
        for (int rt = 0; rt < 4; ++rt) {
          v16h af = load_a_frag(Ab + (size_t)(rb + rt * 16) * 2048 + kk, 2048);
          acc[ct][rt] = wmma32(af, bf, acc[ct][rt]);
        }
      }
    }
    __syncthreads();  // previous phase readers done
#pragma unroll
    for (int ct = 0; ct < 2; ++ct)
#pragma unroll
      for (int rt = 0; rt < 4; ++rt)
        store_c(s_xpk + (rt * 16) * PM + (wave + ct * 8) * 16, PM, acc[ct][rt]);
    __syncthreads();

    int row = threadIdx.x >> 2;
    int j   = threadIdx.x & 3;
    float* xr = s_xpk + row * PM;
    float mx = 0.f;
    for (int c = 0; c < 64; ++c) mx = fmaxf(mx, fabsf(xr[j * 64 + c]));
    mx = fmaxf(mx, __shfl_xor(mx, 1));
    mx = fmaxf(mx, __shfl_xor(mx, 2));
    float s = 0.f;
    for (int c = 0; c < 64; ++c) {
      float v = xr[j * 64 + c];
      s += __expf(v - mx) + __expf(-v - mx);
    }
    s += __shfl_xor(s, 1);
    s += __shfl_xor(s, 2);
    float inv = 1.f / s;
    int r = rb + row, l = r >> 5, b = r & 31;
    float piv = (ph == 0 ? pi0 : pi1)[h * LSEQ + l];
    half_t* dst = kf + (size_t)((b * NH + h) * LSEQ + l) * M2;
    for (int c = 0; c < 64; ++c) {
      float v  = xr[j * 64 + c];
      float fp = __expf(v - mx) * inv * piv;
      float fn = __expf(-v - mx) * inv * piv;
      if (ph == 0) {
        dst[j * 64 + c]      = (half_t)fp;
        dst[PM + j * 64 + c] = (half_t)fn;
      } else {  // same-thread read-modify-write
        dst[j * 64 + c]      = (half_t)((float)dst[j * 64 + c] + fp);
        dst[PM + j * 64 + c] = (half_t)((float)dst[PM + j * 64 + c] + fn);
      }
    }
    __syncthreads();
  }
}

// ---------------- Kernel 4: denom[b,h,l] = qf_l . cumsum_{s<=l} kf_s ----------------
__global__ void k_denom(const half_t* __restrict__ qf, const half_t* __restrict__ kf,
                        float* __restrict__ denom) {
  int bh = blockIdx.x;
  const half_t* qb = qf + (size_t)bh * LSEQ * M2;
  const half_t* kb = kf + (size_t)bh * LSEQ * M2;
  __shared__ float wred[8];
  int t  = threadIdx.x;  // 256
  int m0 = 2 * t;        // each thread owns a consecutive feature pair (b32 loads)
  float a0 = 0.f, a1 = 0.f;
  for (int l = 0; l < LSEQ; ++l) {
    const half_t* kl = kb + l * M2 + m0;
    const half_t* ql = qb + l * M2 + m0;
    a0 += (float)kl[0];
    a1 += (float)kl[1];
    float p = (float)ql[0] * a0 + (float)ql[1] * a1;
#pragma unroll
    for (int m = 16; m >= 1; m >>= 1) p += __shfl_xor(p, m);
    if ((t & 31) == 0) wred[t >> 5] = p;
    __syncthreads();
    if (t == 0) {
      float s = 0.f;
#pragma unroll
      for (int w = 0; w < 8; ++w) s += wred[w];
      denom[bh * LSEQ + l] = s;
    }
    __syncthreads();
  }
}

// ---------------- Kernel 5: causal fast-weight scan, one block per (b,h) ----------------
// Dynamic LDS (294912 B): W32 f32[512][64] | W16T f16[64][512] (transposed mirror) |
//   kbS f16[64][512] (TDM-staged) | vbT f16[64][64] (transposed) | scS f16[64][64] |
//   outS f32[64][64]
__global__ void k_scan(const half_t* __restrict__ qf, const half_t* __restrict__ kf,
                       const half_t* __restrict__ qkv, const float* __restrict__ dn,
                       half_t* __restrict__ attn) {
  extern __shared__ char s_scan[];
  float*  W32  = (float*)s_scan;                       // 131072 B
  half_t* W16T = (half_t*)(s_scan + 131072);           //  65536 B  [d][m]
  half_t* kbS  = (half_t*)(s_scan + 196608);           //  65536 B  [s][m]
  half_t* vbT  = (half_t*)(s_scan + 262144);           //   8192 B  [d][s]
  half_t* scS  = (half_t*)(s_scan + 270336);           //   8192 B  [c][s]
  float*  outS = (float*)(s_scan + 278528);            //  16384 B  [c][d]

  int bh = blockIdx.x, b = bh >> 3, h = bh & 7;
  int wave = threadIdx.x >> 5;
  const half_t* qfb = qf + (size_t)bh * LSEQ * M2;
  const half_t* kfb = kf + (size_t)bh * LSEQ * M2;

  for (int idx = threadIdx.x; idx < M2 * DH; idx += 256) {
    W32[idx]  = 0.f;
    W16T[idx] = (half_t)0.f;
  }
  __syncthreads();

  for (int c4 = 0; c4 < 4; ++c4) {
    int l0 = c4 * CHK;

    // --- TDM: DMA the contiguous 64KB kb chunk (64x512 f16) into LDS ---
    if (wave == 0) {
      unsigned long long ga = (unsigned long long)(uintptr_t)(kfb + (size_t)l0 * M2);
      unsigned lds = (unsigned)(uintptr_t)kbS;
      v4u g0;
      g0[0] = 1u;                                           // count=1
      g0[1] = lds;                                          // lds_addr
      g0[2] = (unsigned)(ga & 0xffffffffu);                 // global_addr[31:0]
      g0[3] = (unsigned)((ga >> 32) & 0x1ffffffu) | 0x80000000u;  // addr[56:32]|type=2
      v8u g1;
      g1[0] = 0x00010000u;          // data_size = 2 bytes
      g1[1] = 0x8000u << 16;        // tensor_dim0 = 32768 (low16 @ bits 63:48)
      g1[2] = 1u << 16;             // tensor_dim1 = 1
      g1[3] = 0x8000u << 16;        // tile_dim0 = 32768 (bits 127:112)
      g1[4] = 0u;                   // tile_dim1 = tile_dim2 = 0
      g1[5] = 32768u;               // tensor_dim0_stride low32
      g1[6] = 0u;
      g1[7] = 0u;
      asm volatile("tensor_load_to_lds %0, %1" :: "s"(g0), "s"(g1) : "memory");
      __builtin_amdgcn_s_wait_tensorcnt(0);
    }
    // stage v transposed [d][s]; zero out accumulator
    for (int idx = threadIdx.x; idx < CHK * DH; idx += 256) {
      int d = idx >> 6, s = idx & 63;
      vbT[idx]  = qkv[(size_t)((l0 + s) * BB + b) * 2048 + h * 256 + 192 + d];
      outS[idx] = 0.f;
    }
    __syncthreads();

    // inter = qb @ W (pre-chunk W16T); wave owns m-slice [64w,64w+64)
    int m0 = wave * 64;
    for (int ct = 0; ct < 4; ++ct)
      for (int rt = 0; rt < 4; ++rt) {
        v8f acc = zero8();
#pragma unroll
        for (int kk = 0; kk < 64; kk += 32) {
          v16h af = load_a_frag(qfb + (size_t)(l0 + rt * 16) * M2 + m0 + kk, M2);
          v16h bf = load_b_frag_T(W16T + (ct * 16) * M2 + m0 + kk, M2);
          acc = wmma32(af, bf, acc);
        }
        atomic_add_c(outS + rt * 16 * DH + ct * 16, DH, acc);
      }
    __syncthreads();

    // sc = tril(qb @ kb^T), stored f16
    for (int tt = 0; tt < 2; ++tt) {
      int tid2 = wave * 2 + tt, rt = tid2 >> 2, ct = tid2 & 3;
      v8f acc = zero8();
      for (int kk = 0; kk < M2; kk += 32) {
        v16h af = load_a_frag(qfb + (size_t)(l0 + rt * 16) * M2 + kk, M2);
        v16h bf = load_b_frag_T(kbS + ct * 16 * M2 + kk, M2);
        acc = wmma32(af, bf, acc);
      }
      {  // masked store
        int lane = threadIdx.x & 31, col = lane & 15, hi = lane >> 4;
#pragma unroll
        for (int v = 0; v < 8; ++v) {
          int cc = rt * 16 + v + hi * 8;
          int ss = ct * 16 + col;
          scS[cc * DH + ss] = (half_t)((ss <= cc) ? acc[v] : 0.f);
        }
      }
    }
    __syncthreads();

    // intra: out += sc @ vb   (vb transposed in LDS)
    for (int tt = 0; tt < 2; ++tt) {
      int tid2 = wave * 2 + tt, rt = tid2 >> 2, ct = tid2 & 3;
      v8f acc = load_c(outS + rt * 16 * DH + ct * 16, DH);
#pragma unroll
      for (int kk = 0; kk < 64; kk += 32) {
        v16h af = load_a_frag(scS + rt * 16 * DH + kk, DH);
        v16h bf = load_b_frag_T(vbT + (ct * 16) * DH + kk, DH);
        acc = wmma32(af, bf, acc);
      }
      store_c(outS + rt * 16 * DH + ct * 16, DH, acc);
    }
    __syncthreads();

    // writeout: attn[l*32+b][h*64+d] = SCALE * out / (denom + eps)
    for (int idx = threadIdx.x; idx < CHK * DH; idx += 256) {
      int cc = idx >> 6, d = idx & 63;
      int l = l0 + cc;
      float val = 0.125f * outS[cc * DH + d] / (dn[bh * LSEQ + l] + 1e-5f);
      attn[(size_t)(l * BB + b) * DMODEL + h * DH + d] = (half_t)val;
    }

    // W += kb^T @ vb (f32 master; mirror transposed to W16T)
    for (int tt = 0; tt < 16; ++tt) {
      int tid2 = wave * 16 + tt;  // 0..127
      int mt = tid2 >> 2, dt = tid2 & 3;
      v8f acc = load_c(W32 + mt * 16 * DH + dt * 16, DH);
#pragma unroll
      for (int kk = 0; kk < 64; kk += 32) {
        v16h af = load_a_frag_T(kbS + kk * M2 + mt * 16, M2);
        v16h bf = load_b_frag_T(vbT + (dt * 16) * DH + kk, DH);
        acc = wmma32(af, bf, acc);
      }
      store_c(W32 + mt * 16 * DH + dt * 16, DH, acc);
      store_c_h_T(W16T + (dt * 16) * M2 + mt * 16, M2, acc);
    }
    __syncthreads();
  }
}

// ---------------- Kernel 6: out = LN(h + attn @ w_o) ----------------
// woT pre-transposed [512][512]; block: 64 rows x 512 cols; wave w owns cols [64w,64w+64).
__global__ void k_oproj_ln(const half_t* __restrict__ attn, const half_t* __restrict__ woT,
                           const float* __restrict__ hsrc, const float* __restrict__ gamma,
                           const float* __restrict__ beta, float* __restrict__ out) {
  extern __shared__ float s_xb[];  // 64*512 f32
  int rb   = blockIdx.x * 64;
  int wave = threadIdx.x >> 5;

  v8f acc[4][4];
#pragma unroll
  for (int r = 0; r < 4; ++r)
#pragma unroll
    for (int j = 0; j < 4; ++j) acc[r][j] = zero8();

  for (int kk = 0; kk < DMODEL; kk += 32) {
    v16h af[4];
#pragma unroll
    for (int rt = 0; rt < 4; ++rt)
      af[rt] = load_a_frag(attn + (size_t)(rb + rt * 16) * DMODEL + kk, DMODEL);
#pragma unroll
    for (int j = 0; j < 4; ++j) {
      int ct = wave * 4 + j;
      v16h bf = load_b_frag_T(woT + (size_t)(ct * 16) * DMODEL + kk, DMODEL);
#pragma unroll
      for (int rt = 0; rt < 4; ++rt) acc[rt][j] = wmma32(af[rt], bf, acc[rt][j]);
    }
  }
#pragma unroll
  for (int rt = 0; rt < 4; ++rt)
#pragma unroll
    for (int j = 0; j < 4; ++j)
      store_c(s_xb + rt * 16 * DMODEL + (wave * 4 + j) * 16, DMODEL, acc[rt][j]);
  __syncthreads();

  // residual + LayerNorm: 4 threads per row, 128 cols each
  int row = threadIdx.x >> 2;
  int j4  = threadIdx.x & 3;
  const float* hrow = hsrc + (size_t)(rb + row) * DMODEL;
  float* xr = s_xb + row * DMODEL;
  float s1 = 0.f, s2 = 0.f;
  for (int c = 0; c < 128; ++c) {
    int col = j4 * 128 + c;
    float v = xr[col] + hrow[col];
    xr[col] = v;
    s1 += v;
    s2 += v * v;
  }
  s1 += __shfl_xor(s1, 1); s1 += __shfl_xor(s1, 2);
  s2 += __shfl_xor(s2, 1); s2 += __shfl_xor(s2, 2);
  float mu  = s1 * (1.f / 512.f);
  float var = s2 * (1.f / 512.f) - mu * mu;
  float inv = rsqrtf(var + 1e-5f);
  float* orow = out + (size_t)(rb + row) * DMODEL;
  for (int c = 0; c < 128; ++c) {
    int col = j4 * 128 + c;
    orow[col] = (xr[col] - mu) * inv * gamma[col] + beta[col];
  }
}

// ---------------- Host launcher ----------------
extern "C" void kernel_launch(void* const* d_in, const int* in_sizes, int n_in,
                              void* d_out, int out_size, void* d_ws, size_t ws_size,
                              hipStream_t stream) {
  const float* h_f    = (const float*)d_in[0];  // [256,32,512]
  const float* wqkv_f = (const float*)d_in[1];  // [512,2048]
  const float* wo_f   = (const float*)d_in[2];  // [512,512]
  const float* gamma  = (const float*)d_in[3];
  const float* beta   = (const float*)d_in[4];
  const float* pi0    = (const float*)d_in[5];  // [8,256]
  const float* pi1    = (const float*)d_in[6];  // [8,256]
  const float* proj_f = (const float*)d_in[7];  // [64,256]
  float* out = (float*)d_out;

  char* ws = (char*)d_ws;
  size_t off = 0;
  auto alloc = [&](size_t bytes) {
    size_t o = off;
    off += (bytes + 255) & ~(size_t)255;
    return o;
  };
  half_t* h16     = (half_t*)(ws + alloc((size_t)ROWS * DMODEL * 2));
  half_t* wqkvT   = (half_t*)(ws + alloc((size_t)DMODEL * 2048 * 2));   // [2048][512]
  half_t* woT     = (half_t*)(ws + alloc((size_t)DMODEL * DMODEL * 2)); // [512][512]
  half_t* projT   = (half_t*)(ws + alloc((size_t)DH * PM * 2));         // [256][64]
  half_t* proj2T  = (half_t*)(ws + alloc((size_t)DH * PM * 2));         // [256][64]
  half_t* qkv16   = (half_t*)(ws + alloc((size_t)ROWS * 2048 * 2));
  half_t* qf16    = (half_t*)(ws + alloc((size_t)BB * NH * LSEQ * M2 * 2));
  half_t* kf16    = (half_t*)(ws + alloc((size_t)BB * NH * LSEQ * M2 * 2));
  float*  denom   = (float*)(ws + alloc((size_t)BB * NH * LSEQ * 4));
  half_t* attn16  = (half_t*)(ws + alloc((size_t)ROWS * DMODEL * 2));
  (void)ws_size; (void)n_in; (void)in_sizes; (void)out_size;

  const float XSCALE = 0.3535533906f;  // 64^-0.25, folded into proj

  auto cast = [&](const float* s, half_t* d, int n, float sc) {
    int blocks = (n + 255) / 256;
    if (blocks > 4096) blocks = 4096;
    k_cast_f16<<<blocks, 256, 0, stream>>>(s, d, n, sc);
  };
  auto castT = [&](const float* s, half_t* d, int R, int C, float sc) {
    int n = R * C;
    int blocks = (n + 255) / 256;
    if (blocks > 4096) blocks = 4096;
    k_cast_t16<<<blocks, 256, 0, stream>>>(s, d, R, C, sc);
  };
  cast(h_f, h16, ROWS * DMODEL, 1.f);
  castT(wqkv_f, wqkvT, DMODEL, 2048, 1.f);
  castT(wo_f, woT, DMODEL, DMODEL, 1.f);
  castT(proj_f, projT, DH, PM, XSCALE);
  castT(proj_f, proj2T, DH, PM, 0.7f * XSCALE);

  k_gemm_qkv<<<dim3(ROWS / 64, 2048 / 128), 256, 0, stream>>>(h16, wqkvT, qkv16);
  k_prime_q<<<dim3(ROWS / 64, NH), 256, 64 * PM * 4, stream>>>(qkv16, projT, qf16);
  k_prime_k<<<dim3(ROWS / 64, NH), 256, 64 * PM * 4, stream>>>(qkv16, projT, proj2T,
                                                               pi0, pi1, kf16);
  k_denom<<<BB * NH, 256, 0, stream>>>(qf16, kf16, denom);
  k_scan<<<BB * NH, 256, 294912, stream>>>(qf16, kf16, qkv16, denom, attn16);
  k_oproj_ln<<<ROWS / 64, 256, 64 * DMODEL * 4, stream>>>(attn16, woT, h_f, gamma,
                                                          beta, out);
}
